// LSTMHeteroPost_50766513439427
// MI455X (gfx1250) — matrix operand order repaired and checked by reading, not compiled, not run on hardware
//
#include <hip/hip_runtime.h>

// ---------------------------------------------------------------------------
// CDNA5 / gfx1250: wave32, WMMA bf16 16x16x32, double-buffered LDS h-exchange,
// split s_barrier_signal/_wait with VMEM overlap, native bf16 converts,
// branch-free transcendentals (v_exp_f32 / v_rcp_f32), global prefetch.
// ---------------------------------------------------------------------------

typedef __attribute__((ext_vector_type(16))) __bf16 v16bf;
typedef __attribute__((ext_vector_type(8)))  float  v8f;

union BF16Frag {
    v16bf v;
    unsigned short s[16];
    unsigned int   u[8];
};

// native RNE f32->bf16 convert (hardware cvt on gfx1250)
__device__ __forceinline__ unsigned short f2bf(float f) {
    union { __bf16 h; unsigned short u; } c;
    c.h = (__bf16)f;
    return c.u;
}

__device__ __forceinline__ unsigned int pack2(float a, float b) {
    union { __bf16 h[2]; unsigned int u; } c;
    c.h[0] = (__bf16)a;
    c.h[1] = (__bf16)b;
    return c.u;
}

// branch-free sigmoid: v_exp_f32 + v_rcp_f32 (co-executes with XDL WMMA)
__device__ __forceinline__ float sigm(float x) {
    return __builtin_amdgcn_rcpf(1.0f + __expf(-x));
}

// branch-free tanh: 1 - 2/(exp(2x)+1); exp overflow->inf gives exact +/-1
__device__ __forceinline__ float fast_tanh(float x) {
    float e = __expf(2.0f * x);
    return 1.0f - 2.0f * __builtin_amdgcn_rcpf(e + 1.0f);
}

// D = A(16x32 bf16) * B(32x16 bf16) + C(16x16 f32)
__device__ __forceinline__ v8f wmma_bf16(const BF16Frag& a, const BF16Frag& b, v8f c) {
    return __builtin_amdgcn_wmma_f32_16x16x32_bf16(
        /*neg_a=*/false, a.v, /*neg_b=*/false, b.v,
        /*c_mod=*/(short)0, c, /*reuse_a=*/false, /*reuse_b=*/false);
}

// B-fragment for B = W^T (W row-major [N, K]): lane ln holds column n,
// element j <-> K = k0 + j  (k0 = kt*32 + lg*16). 16 consecutive floats.
__device__ __forceinline__ BF16Frag load_bfrag(const float* __restrict__ W,
                                               int ldW, int n, int k0) {
    const float* p = W + (size_t)n * ldW + k0;
    float4 f0 = *(const float4*)(p);
    float4 f1 = *(const float4*)(p + 4);
    float4 f2 = *(const float4*)(p + 8);
    float4 f3 = *(const float4*)(p + 12);
    BF16Frag b;
    b.u[0] = pack2(f0.x, f0.y); b.u[1] = pack2(f0.z, f0.w);
    b.u[2] = pack2(f1.x, f1.y); b.u[3] = pack2(f1.z, f1.w);
    b.u[4] = pack2(f2.x, f2.y); b.u[5] = pack2(f2.z, f2.w);
    b.u[6] = pack2(f3.x, f3.y); b.u[7] = pack2(f3.z, f3.w);
    return b;
}

// A-fragment from a global f32 row (K-major): element j <-> K=kt*32+16*lg+j.
__device__ __forceinline__ BF16Frag load_afrag_g(const float* __restrict__ row,
                                                 int kt, int lg) {
    const float* p = row + kt * 32 + lg * 16;
    float4 f0 = *(const float4*)(p);
    float4 f1 = *(const float4*)(p + 4);
    float4 f2 = *(const float4*)(p + 8);
    float4 f3 = *(const float4*)(p + 12);
    BF16Frag a;
    a.u[0] = pack2(f0.x, f0.y); a.u[1] = pack2(f0.z, f0.w);
    a.u[2] = pack2(f1.x, f1.y); a.u[3] = pack2(f1.z, f1.w);
    a.u[4] = pack2(f2.x, f2.y); a.u[5] = pack2(f2.z, f2.w);
    a.u[6] = pack2(f3.x, f3.y); a.u[7] = pack2(f3.z, f3.w);
    return a;
}

// A-fragment for the x tile: K = F = 16 valid, K=16..31 zero-padded.
__device__ __forceinline__ BF16Frag load_afrag_x(const float* __restrict__ row, int lg) {
    BF16Frag a;
    if (lg == 0) {
        float4 f0 = *(const float4*)(row);
        float4 f1 = *(const float4*)(row + 4);
        float4 f2 = *(const float4*)(row + 8);
        float4 f3 = *(const float4*)(row + 12);
        a.u[0] = pack2(f0.x, f0.y); a.u[1] = pack2(f0.z, f0.w);
        a.u[2] = pack2(f1.x, f1.y); a.u[3] = pack2(f1.z, f1.w);
        a.u[4] = pack2(f2.x, f2.y); a.u[5] = pack2(f2.z, f2.w);
        a.u[6] = pack2(f3.x, f3.y); a.u[7] = pack2(f3.z, f3.w);
    } else {
        for (int j = 0; j < 8; ++j) a.u[j] = 0u;
    }
    return a;
}

// A-fragment from an LDS bf16 buffer with padded row stride 136 ushorts (272 B,
// 16B-aligned + rotates LDS banks across rows).  Two ds_load_b128 per fragment.
#define LDS_LD 136
__device__ __forceinline__ BF16Frag load_afrag_lds(const unsigned short* buf,
                                                   int m, int kt, int lg) {
    const unsigned short* p = buf + m * LDS_LD + kt * 32 + lg * 16;
    uint4 lo = *(const uint4*)(p);
    uint4 hi = *(const uint4*)(p + 8);
    BF16Frag a;
    a.u[0] = lo.x; a.u[1] = lo.y; a.u[2] = lo.z; a.u[3] = lo.w;
    a.u[4] = hi.x; a.u[5] = hi.y; a.u[6] = hi.z; a.u[7] = hi.w;
    return a;
}

// ---------------------------------------------------------------------------
// Kernel 1: fused LSTM over T=256 steps + mean over the 16 agents of batch b.
// One workgroup per b; 8 waves; wave w owns columns {g*128 + 16w .. +16} of
// the 512 gate outputs -> cell update is wave-local.  Whh^T/Wih^T fragments
// are register-resident; h ping-pongs through LDS (bf16).  One split barrier
// per step with the next x-tile VMEM load issued between signal and wait.
// ---------------------------------------------------------------------------
__global__ __launch_bounds__(256, 1) void lstm_mean_kernel(
    const float* __restrict__ agent_obs,   // [512, 256, 16, 16]
    const float* __restrict__ Wih,         // [512, 16]
    const float* __restrict__ Whh,         // [512, 128]
    const float* __restrict__ bih,         // [512]
    const float* __restrict__ bhh,         // [512]
    float* __restrict__ mean_a)            // [512, 128]
{
    __shared__ __align__(16) unsigned short h_buf[2][16 * LDS_LD];

    const int tid  = threadIdx.x;
    const int wave = tid >> 5;
    const int lane = tid & 31;
    const int lg   = lane >> 4;   // lane group 0/1
    const int ln   = lane & 15;   // n within tile (B/C/D) and m (A)
    const int b    = blockIdx.x;

    // ---- persistent weight fragments + bias splat vectors ----
    BF16Frag Bh[4][4];            // Whh^T: [gate][ktile]
    BF16Frag Bx[4];               // Wih^T: [gate]
    v8f      biasv[4];            // per-gate bias broadcast, used as WMMA C
#pragma unroll
    for (int g = 0; g < 4; ++g) {
        const int n = g * 128 + wave * 16 + ln;
#pragma unroll
        for (int kt = 0; kt < 4; ++kt)
            Bh[g][kt] = load_bfrag(Whh, 128, n, kt * 32 + lg * 16);
        if (lg == 0) {
            Bx[g] = load_bfrag(Wih, 16, n, 0);  // K rows 0..15
        } else {                                 // K rows 16..31 multiply zeros
#pragma unroll
            for (int j = 0; j < 8; ++j) Bx[g].u[j] = 0u;
        }
        const float bg = bih[n] + bhh[n];
#pragma unroll
        for (int r = 0; r < 8; ++r) biasv[g][r] = bg;
    }

    for (int i = tid; i < 16 * LDS_LD; i += 256) h_buf[0][i] = 0;

    v8f c_state, h_tile;
#pragma unroll
    for (int r = 0; r < 8; ++r) { c_state[r] = 0.0f; h_tile[r] = 0.0f; }

    __syncthreads();

    const float* xbase = agent_obs + (size_t)b * 256 * 256;  // T*N*F floats
    int cur = 0;                                             // ping-pong index

    BF16Frag Ax = load_afrag_x(xbase + ln * 16, lg);         // t = 0 tile

    for (int t = 0; t < 256; ++t) {
        v8f acc[4];
#pragma unroll
        for (int g = 0; g < 4; ++g)
            acc[g] = wmma_bf16(Ax, Bx[g], biasv[g]);         // x @ Wih^T + b
#pragma unroll
        for (int kt = 0; kt < 4; ++kt) {                     // h(t-1) @ Whh^T
            BF16Frag Ah = load_afrag_lds(h_buf[cur], ln, kt, lg);
#pragma unroll
            for (int g = 0; g < 4; ++g)
                acc[g] = wmma_bf16(Ah, Bh[g][kt], acc[g]);
        }

        // wave-local LSTM cell update (gate order i,f,g,o); write h(t) into
        // the OTHER buffer -> one split barrier per step orders everything.
#pragma unroll
        for (int r = 0; r < 8; ++r) {
            float iv = sigm(acc[0][r]);
            float fv = sigm(acc[1][r]);
            float gv = fast_tanh(acc[2][r]);
            float ov = sigm(acc[3][r]);
            float c  = fv * c_state[r] + iv * gv;
            c_state[r] = c;
            float h  = ov * fast_tanh(c);
            h_tile[r] = h;
            h_buf[cur ^ 1][(r + 8 * lg) * LDS_LD + wave * 16 + ln] = f2bf(h);
        }

        // split barrier: drain LDS stores, signal, overlap the next x-tile
        // global load + prefetch with the barrier latency, then wait.
        asm volatile("s_wait_dscnt 0x0" ::: "memory");
        asm volatile("s_barrier_signal -1" ::: "memory");
        if (t + 1 < 256) {
            Ax = load_afrag_x(xbase + (t + 1) * 256 + ln * 16, lg);
            __builtin_prefetch(xbase + (t + 2) * 256 + ln * 16 + lg * 8, 0, 0);
        }
        asm volatile("s_barrier_wait -1" ::: "memory");
        cur ^= 1;
    }

    // mean over the 16 agents (rows of the D tile): sum 8 register rows,
    // then fold the other lane group (m 8..15 vs 0..7, same column).
    float part = 0.0f;
#pragma unroll
    for (int r = 0; r < 8; ++r) part += h_tile[r];
    part += __shfl_xor(part, 16, 32);
    if (lg == 0)
        mean_a[b * 128 + wave * 16 + ln] = part * (1.0f / 16.0f);
}

// ---------------------------------------------------------------------------
// Kernel 2: collapsed hetero-GNN epilogue.  Only o_ss of layer 2 feeds the
// output; everything reduces to three [.,128]@[128,128] WMMA GEMMs, one K=2
// VALU GEMM, constant vectors, tanh, and the concat tail.
// 32 blocks x 16 rows; wave w owns output columns 16w..16w+16.
// ---------------------------------------------------------------------------
__global__ __launch_bounds__(256, 1) void gnn_out_kernel(
    const float* __restrict__ mean_a,      // [512,128]
    const float* __restrict__ hideout,     // [512,2]
    const float* __restrict__ timestep,    // [512,1]
    const float* __restrict__ Wl1_a2as,    // [128,128]
    const float* __restrict__ bl1_a2as,    // [128]
    const float* __restrict__ bl1_ss2as,   // [128]
    const float* __restrict__ Wl1_h2hs,    // [128,2]
    const float* __restrict__ bl1_h2hs,    // [128]
    const float* __restrict__ bl1_ss2hs,   // [128]
    const float* __restrict__ bl1_hs2ss,   // [128]
    const float* __restrict__ bl1_as2ss,   // [128]
    const float* __restrict__ Wl2_hs2ss,   // [128,128]
    const float* __restrict__ bl2_hs2ss,   // [128]
    const float* __restrict__ Wr2_hs2ss,   // [128,128]
    const float* __restrict__ Wl2_as2ss,   // [128,128]
    const float* __restrict__ bl2_as2ss,   // [128]
    const float* __restrict__ Wr2_as2ss,   // [128,128]
    float* __restrict__ out)               // [512,131]
{
    __shared__ __align__(16) unsigned short as1[16 * LDS_LD];
    __shared__ __align__(16) unsigned short hs1[16 * LDS_LD];

    const int tid  = threadIdx.x;
    const int wave = tid >> 5;
    const int lane = tid & 31;
    const int lg   = lane >> 4;
    const int ln   = lane & 15;
    const int m0   = blockIdx.x * 16;
    const int n    = wave * 16 + ln;

    // ---- stage A1: o_as1 = mean_a @ Wl1_a2as^T + bl1_a2as + bl1_ss2as ----
    {
        v8f acc;
        const float bias = bl1_a2as[n] + bl1_ss2as[n];
#pragma unroll
        for (int r = 0; r < 8; ++r) acc[r] = bias;
#pragma unroll
        for (int kt = 0; kt < 4; ++kt) {
            BF16Frag A = load_afrag_g(mean_a + (size_t)(m0 + ln) * 128, kt, lg);
            BF16Frag B = load_bfrag(Wl1_a2as, 128, n, kt * 32 + lg * 16);
            acc = wmma_bf16(A, B, acc);
        }
#pragma unroll
        for (int r = 0; r < 8; ++r)
            as1[(r + 8 * lg) * LDS_LD + n] = f2bf(acc[r]);
    }

    // ---- stage A2: o_hs1 = hideout @ Wl1_h2hs^T + bl1_h2hs + bl1_ss2hs ----
    for (int idx = tid; idx < 16 * 128; idx += 256) {
        const int m  = idx >> 7;
        const int nn = idx & 127;
        const int row = m0 + m;
        float v = hideout[row * 2 + 0] * Wl1_h2hs[nn * 2 + 0]
                + hideout[row * 2 + 1] * Wl1_h2hs[nn * 2 + 1]
                + bl1_h2hs[nn] + bl1_ss2hs[nn];
        hs1[m * LDS_LD + nn] = f2bf(v);
    }
    __syncthreads();

    // ---- stage B: o_ss2 ----
    // constant part: biases + o_ss1 @ (Wr2_hs2ss + Wr2_as2ss)^T (row-indep.)
    float cconst = bl2_hs2ss[n] + bl2_as2ss[n];
    for (int k = 0; k < 128; ++k) {
        float ss1k = bl1_hs2ss[k] + bl1_as2ss[k];
        cconst += ss1k * (Wr2_hs2ss[n * 128 + k] + Wr2_as2ss[n * 128 + k]);
    }
    v8f acc2;
#pragma unroll
    for (int r = 0; r < 8; ++r) acc2[r] = cconst;
#pragma unroll
    for (int kt = 0; kt < 4; ++kt) {
        BF16Frag Ah = load_afrag_lds(hs1, ln, kt, lg);
        BF16Frag Bh = load_bfrag(Wl2_hs2ss, 128, n, kt * 32 + lg * 16);
        acc2 = wmma_bf16(Ah, Bh, acc2);
        BF16Frag Aa = load_afrag_lds(as1, ln, kt, lg);
        BF16Frag Ba = load_bfrag(Wl2_as2ss, 128, n, kt * 32 + lg * 16);
        acc2 = wmma_bf16(Aa, Ba, acc2);
    }

    // ---- store: tanh(o_ss2) into cols 0..127 ----
#pragma unroll
    for (int r = 0; r < 8; ++r) {
        const int row = m0 + r + 8 * lg;
        out[(size_t)row * 131 + n] = fast_tanh(acc2[r]);
    }
    // ---- concat tail: hideout (2) + timestep (1) ----
    if (tid < 16) {
        const int row = m0 + tid;
        out[(size_t)row * 131 + 128] = hideout[row * 2 + 0];
        out[(size_t)row * 131 + 129] = hideout[row * 2 + 1];
        out[(size_t)row * 131 + 130] = timestep[row];
    }
}

// ---------------------------------------------------------------------------
// Input flattening (setup_inputs dict order, leaves depth-first):
//  0 agent_obs  1 hideout  2 timestep  3 num_agents
//  4 Wih  5 Whh  6 bih  7 bhh
//  conv1 (Wl,bl,Wr per relation): a2as 8-10, h2hs 11-13, hs2ss 14-16,
//    as2ss 17-19, as2a 20-22, hs2h 23-25, ss2hs 26-28, ss2as 29-31
//  conv2: a2as 32-34, h2hs 35-37, hs2ss 38-40, as2ss 41-43, as2a 44-46,
//    hs2h 47-49, ss2hs 50-52, ss2as 53-55
// ---------------------------------------------------------------------------
extern "C" void kernel_launch(void* const* d_in, const int* in_sizes, int n_in,
                              void* d_out, int out_size, void* d_ws, size_t ws_size,
                              hipStream_t stream) {
    const float* agent_obs = (const float*)d_in[0];
    const float* hideout   = (const float*)d_in[1];
    const float* timestep  = (const float*)d_in[2];
    const float* Wih       = (const float*)d_in[4];
    const float* Whh       = (const float*)d_in[5];
    const float* bih       = (const float*)d_in[6];
    const float* bhh       = (const float*)d_in[7];

    const float* Wl1_a2as  = (const float*)d_in[8];
    const float* bl1_a2as  = (const float*)d_in[9];
    const float* Wl1_h2hs  = (const float*)d_in[11];
    const float* bl1_h2hs  = (const float*)d_in[12];
    const float* bl1_hs2ss = (const float*)d_in[15];
    const float* bl1_as2ss = (const float*)d_in[18];
    const float* bl1_ss2hs = (const float*)d_in[27];
    const float* bl1_ss2as = (const float*)d_in[30];

    const float* Wl2_hs2ss = (const float*)d_in[38];
    const float* bl2_hs2ss = (const float*)d_in[39];
    const float* Wr2_hs2ss = (const float*)d_in[40];
    const float* Wl2_as2ss = (const float*)d_in[41];
    const float* bl2_as2ss = (const float*)d_in[42];
    const float* Wr2_as2ss = (const float*)d_in[43];

    float* mean_a = (float*)d_ws;          // 512*128 f32 = 256 KB scratch
    float* out    = (float*)d_out;         // [512,131]

    (void)in_sizes; (void)n_in; (void)out_size; (void)ws_size;

    lstm_mean_kernel<<<512, 256, 0, stream>>>(agent_obs, Wih, Whh, bih, bhh, mean_a);
    gnn_out_kernel<<<32, 256, 0, stream>>>(mean_a, hideout, timestep,
                                           Wl1_a2as, bl1_a2as, bl1_ss2as,
                                           Wl1_h2hs, bl1_h2hs, bl1_ss2hs,
                                           bl1_hs2ss, bl1_as2ss,
                                           Wl2_hs2ss, bl2_hs2ss, Wr2_hs2ss,
                                           Wl2_as2ss, bl2_as2ss, Wr2_as2ss,
                                           out);
}
